// GNN_86586540687361
// MI455X (gfx1250) — matrix operand order repaired and checked by reading, not compiled, not run on hardware
//
#include <hip/hip_runtime.h>

#define NN 100000
#define NE 1600000

typedef __attribute__((ext_vector_type(16))) __bf16 v16bf;
typedef __attribute__((ext_vector_type(8)))  __bf16 v8bf;
typedef __attribute__((ext_vector_type(8)))  float  v8f;

// Order-preserving float-bits -> uint bijection (needed only for mixed-sign
// layer 1; post-ReLU layers use raw bits, which are already monotone >= 0).
static __device__ __forceinline__ unsigned encBits(unsigned b) {
  return (b & 0x80000000u) ? ~b : (b | 0x80000000u);
}
static __device__ __forceinline__ unsigned decBits(unsigned u) {
  if (u == 0u) return 0u;  // empty segment -> 0 (matches isfinite fixup)
  return (u & 0x80000000u) ? (u ^ 0x80000000u) : ~u;
}
static __device__ __forceinline__ __bf16 f2bf(float f) {
  unsigned u = __builtin_bit_cast(unsigned, f);
  u += 0x7FFFu + ((u >> 16) & 1u);  // round-to-nearest-even
  unsigned short s = (unsigned short)(u >> 16);
  return __builtin_bit_cast(__bf16, s);
}
struct v16pair { v8bf lo, hi; };
struct u8pack  { unsigned q[8]; };

// -------- f32 -> bf16 row-major convert (weights + input x, once per launch) ----
__global__ void __launch_bounds__(256) cvt_bf(const float* __restrict__ src,
                                              __bf16* __restrict__ dst, int n) {
  int i = blockIdx.x * 256 + (int)threadIdx.x;
  if (i < n) dst[i] = f2bf(src[i]);
}

// -------- scatter: agg[dst] = max(agg[dst], h[src]); h is bf16 ------------------
// Lane covers 8 features (16B load). ENC=false (post-ReLU, values >= 0): atomic
// max on raw f32 bits; init 0 == +0.0 is exact since true max >= 0, and empty
// segments decode to 0.0. ENC=true (mixed sign, layer 1): ordered bijection.
template<int DIN, bool ENC>
__global__ void __launch_bounds__(256) sage_scatter(
    const __bf16* __restrict__ h, const int* __restrict__ ei,
    unsigned* __restrict__ agg) {
  constexpr int LPE = DIN / 8;  // lanes per edge (power of two)
  long long tid = (long long)blockIdx.x * 256 + threadIdx.x;
  long long e = tid / LPE;
  if (e >= NE) return;
  int sub = (int)(tid - e * LPE);
  long long s = ei[e];
  long long d = ei[NE + e];
  uint4 v = *(const uint4*)((const unsigned short*)h + s * DIN + sub * 8);
  unsigned* p = agg + d * DIN + sub * 8;
  unsigned w[4] = {v.x, v.y, v.z, v.w};
#pragma unroll
  for (int j = 0; j < 4; ++j) {
    unsigned lo = w[j] << 16;
    unsigned hi = w[j] & 0xFFFF0000u;
    if (ENC) { lo = encBits(lo); hi = encBits(hi); }
    atomicMax(p + 2 * j + 0, lo);
    atomicMax(p + 2 * j + 1, hi);
  }
}

// -------- fused GEMM: out = [relu]( dec(agg)@Wl^T + b + h@Wr^T ) ----------------
// One wave -> 16 nodes x 64 outputs, v_wmma_f32_16x16x32_bf16, K over [agg|h].
// h and weights are bf16; agg holds (encoded) bits of exact-bf16 values.
template<int DIN, bool ENC, bool RELU, bool F32OUT>
__global__ void __launch_bounds__(256) sage_gemm(
    const unsigned* __restrict__ agg, const __bf16* __restrict__ h,
    const __bf16* __restrict__ Wl, const float* __restrict__ bias,
    const __bf16* __restrict__ Wr, void* __restrict__ outv) {
  const int wave = (int)((blockIdx.x * 256u + threadIdx.x) >> 5);
  if (wave >= NN / 16) return;  // wave-uniform: EXEC all-ones for WMMA
  const int lane = threadIdx.x & 31;
  const int m    = lane & 15;
  const int half = lane >> 4;
  const int row0 = wave * 16;

  const long long arow = (long long)(row0 + m) * DIN;
  const __bf16*   hr = h   + arow;
  const unsigned* ur = agg + arow;

  v8f acc[4];
#pragma unroll
  for (int nt = 0; nt < 4; ++nt) {
    const float bv = bias[nt * 16 + m];
#pragma unroll
    for (int r = 0; r < 8; ++r) acc[nt][r] = bv;
  }

#pragma unroll
  for (int p = 0; p < 2; ++p) {
    const __bf16* W = p ? Wr : Wl;
#pragma unroll
    for (int kb = 0; kb < DIN; kb += 32) {
      // ---- A: 16x32 bf16 tile. Lane holds row m, K = ka+{0..7} and ka+16..23.
      const int ka = kb + half * 8;
      v16bf A;
      if (p == 0) {
        uint4 u0 = *(const uint4*)(ur + ka);
        uint4 u1 = *(const uint4*)(ur + ka + 4);
        uint4 u2 = *(const uint4*)(ur + ka + 16);
        uint4 u3 = *(const uint4*)(ur + ka + 20);
        unsigned du[16] = {u0.x, u0.y, u0.z, u0.w, u1.x, u1.y, u1.z, u1.w,
                           u2.x, u2.y, u2.z, u2.w, u3.x, u3.y, u3.z, u3.w};
        if (ENC) {
#pragma unroll
          for (int j = 0; j < 16; ++j) du[j] = decBits(du[j]);
        }
        u8pack up;  // values are exact bf16<<16: pack high halves, no rounding
#pragma unroll
        for (int j = 0; j < 8; ++j)
          up.q[j] = (du[2 * j] >> 16) | (du[2 * j + 1] & 0xFFFF0000u);
        A = __builtin_bit_cast(v16bf, up);
      } else {
        v16pair t;
        t.lo = *(const v8bf*)(hr + ka);        // K = ka+0..7
        t.hi = *(const v8bf*)(hr + ka + 16);   // K = ka+16..23
        A = __builtin_bit_cast(v16bf, t);
      }

      // ---- B: 32x16 bf16 tile; B[k][n] = W[n][k]. Lane holds col m,
      // K = kw+{0..7} and kw+16..23, kw = kb + 8*half (2x b128 loads).
      const int kw = kb + half * 8;
#pragma unroll
      for (int nt = 0; nt < 4; ++nt) {
        const __bf16* wp = W + (long long)(nt * 16 + m) * DIN + kw;
        v16pair tb;
        tb.lo = *(const v8bf*)(wp);
        tb.hi = *(const v8bf*)(wp + 16);
        v16bf B = __builtin_bit_cast(v16bf, tb);
        acc[nt] = __builtin_amdgcn_wmma_f32_16x16x32_bf16(
            false, A, false, B, (short)0, acc[nt], false, false);
      }
    }
  }

  // D layout: VGPR r -> row r + 8*half, col = nt*16 + m
#pragma unroll
  for (int nt = 0; nt < 4; ++nt) {
    const int n = nt * 16 + m;
#pragma unroll
    for (int r = 0; r < 8; ++r) {
      float v = acc[nt][r];
      if (RELU) v = fmaxf(v, 0.0f);
      const long long idx = (long long)(row0 + half * 8 + r) * 64 + n;
      if (F32OUT) ((float*)outv)[idx]  = v;
      else        ((__bf16*)outv)[idx] = f2bf(v);
    }
  }
}

extern "C" void kernel_launch(void* const* d_in, const int* in_sizes, int n_in,
                              void* d_out, int out_size, void* d_ws, size_t ws_size,
                              hipStream_t stream) {
  (void)in_sizes; (void)n_in; (void)out_size; (void)ws_size;
  const float* x   = (const float*)d_in[0];
  const int*   ei  = (const int*)  d_in[1];
  const float* W1l = (const float*)d_in[2];
  const float* b1  = (const float*)d_in[3];
  const float* W1r = (const float*)d_in[4];
  const float* Wml = (const float*)d_in[5];
  const float* bm  = (const float*)d_in[6];
  const float* Wmr = (const float*)d_in[7];
  const float* W6l = (const float*)d_in[8];
  const float* b6  = (const float*)d_in[9];
  const float* W6r = (const float*)d_in[10];

  char*     ws   = (char*)d_ws;
  unsigned* agg  = (unsigned*)ws;                              // NN*128*4 B
  __bf16*   xb   = (__bf16*)(ws + (size_t)NN * 128 * 4);       // NN*128*2 B
  __bf16*   ping = xb   + (size_t)NN * 128;                    // NN*64*2 B
  __bf16*   pong = ping + (size_t)NN * 64;                     // NN*64*2 B
  __bf16*   w1l  = pong + (size_t)NN * 64;
  __bf16*   w1r  = w1l + 8192;   // 64*128
  __bf16*   wml  = w1r + 8192;   // 4*64*64
  __bf16*   wmr  = wml + 16384;  // 4*64*64
  __bf16*   w6l  = wmr + 16384;  // 64*64
  __bf16*   w6r  = w6l + 4096;   // 64*64

  // one-shot conversions (re-done every call for determinism)
  cvt_bf<<<(NN * 128 + 255) / 256, 256, 0, stream>>>(x, xb, NN * 128);
  cvt_bf<<<(8192  + 255) / 256, 256, 0, stream>>>(W1l, w1l, 8192);
  cvt_bf<<<(8192  + 255) / 256, 256, 0, stream>>>(W1r, w1r, 8192);
  cvt_bf<<<(16384 + 255) / 256, 256, 0, stream>>>(Wml, wml, 16384);
  cvt_bf<<<(16384 + 255) / 256, 256, 0, stream>>>(Wmr, wmr, 16384);
  cvt_bf<<<(4096  + 255) / 256, 256, 0, stream>>>(W6l, w6l, 4096);
  cvt_bf<<<(4096  + 255) / 256, 256, 0, stream>>>(W6r, w6r, 4096);

  const int gemmBlocks = (NN / 16 + 7) / 8;  // 8 waves per 256-thread block

  // layer 1: 128 -> 64, mixed-sign input -> encoded atomics; ReLU, bf16 out
  hipMemsetAsync(agg, 0, (size_t)NN * 128 * 4, stream);
  sage_scatter<128, true><<<(NE * 16) / 256, 256, 0, stream>>>(xb, ei, agg);
  sage_gemm<128, true, true, false><<<gemmBlocks, 256, 0, stream>>>(
      agg, xb, w1l, b1, w1r, ping);

  // layers 2..5: 64 -> 64, inputs >= 0 -> raw-bit atomics; ReLU, bf16 out
  __bf16* cur = ping; __bf16* nxt = pong;
  for (int i = 0; i < 4; ++i) {
    hipMemsetAsync(agg, 0, (size_t)NN * 64 * 4, stream);
    sage_scatter<64, false><<<(NE * 8) / 256, 256, 0, stream>>>(cur, ei, agg);
    sage_gemm<64, false, true, false><<<gemmBlocks, 256, 0, stream>>>(
        agg, cur, wml + (size_t)i * 64 * 64, bm + (size_t)i * 64,
        wmr + (size_t)i * 64 * 64, nxt);
    __bf16* t = cur; cur = nxt; nxt = t;
  }

  // layer 6: 64 -> 64, no ReLU, f32 out to d_out
  hipMemsetAsync(agg, 0, (size_t)NN * 64 * 4, stream);
  sage_scatter<64, false><<<(NE * 8) / 256, 256, 0, stream>>>(cur, ei, agg);
  sage_gemm<64, false, false, true><<<gemmBlocks, 256, 0, stream>>>(
      agg, cur, w6l, b6, w6r, d_out);
}
